// DistanceEdgePredictor_80333068304700
// MI455X (gfx1250) — matrix operand order repaired and checked by reading, not compiled, not run on hardware
//
#include <hip/hip_runtime.h>

// ---------------------------------------------------------------------------
// DistanceEdgePredictor for MI455X (gfx1250, wave32, WMMA).
//
// Pipeline:
//   K1 encode_kernel : h = relu(X @ W^T + b)  [N,64] fp32, split into bf16
//                      hi/lo arrays + sq[n] = ||h_n||^2 (wave32 shfl reduce).
//   K2 pair_kernel   : per 16x16 tile of G = h h^T, 6x v_wmma_f32_16x16x32_bf16
//                      (hi*hi + hi*lo + lo*hi over two K=32 blocks -> fp32-class
//                      accuracy), then d2 = sq_i + sq_j - 2G, dist, sigmoid,
//                      write probs (f32) + pair_index (2x int64) with NT stores.
// Store-bandwidth bound: ~671 MB of outputs -> ~29us floor @ 23.3 TB/s.
// ---------------------------------------------------------------------------

#define NODE_DIM   128
#define HIDDEN_DIM 64

typedef __attribute__((ext_vector_type(16))) __bf16 v16bf;
typedef __attribute__((ext_vector_type(8)))  __bf16 v8bf;
typedef __attribute__((ext_vector_type(8)))  float  v8f;

union V16U {
    v16bf v;
    v8bf  h[2];
};

__device__ __forceinline__ unsigned short f32_to_bf16_rne(float f) {
    unsigned u = __float_as_uint(f);
    unsigned r = u + 0x7FFFu + ((u >> 16) & 1u);   // round-to-nearest-even
    return (unsigned short)(r >> 16);
}
__device__ __forceinline__ float bf16_bits_to_f32(unsigned short s) {
    return __uint_as_float(((unsigned)s) << 16);
}

// sigmoid(thr - dist) with fast HW transcendentals (v_sqrt/v_exp/v_rcp, ~1ulp:
// far more accurate than the bf16-split Gram feeding it).
__device__ __forceinline__ float edge_prob(float g, float si, float sj, float thr) {
    float d2   = fmaf(-2.0f, g, si + sj);
    float dist = __builtin_amdgcn_sqrtf(fmaxf(d2, 0.0f));
    float e    = __expf(dist - thr);               // v_exp_f32 path
    return __builtin_amdgcn_rcpf(1.0f + e);        // v_rcp_f32
}

// ---------------------------------------------------------------------------
// Kernel 1: encoder. One wave (32 lanes) per node row; each lane computes two
// hidden dims (d, d+32) -> 256 FMAs/lane, then wave32 shfl_xor reduction for
// ||h||^2. Emits bf16 hi/lo split for the WMMA Gram stage.
// ---------------------------------------------------------------------------
__global__ void encode_kernel(const float* __restrict__ X,
                              const float* __restrict__ W,
                              const float* __restrict__ bias,
                              unsigned short* __restrict__ hhi,
                              unsigned short* __restrict__ hlo,
                              float* __restrict__ sq,
                              int N) {
    const int lane = threadIdx.x & 31;
    const int wid  = threadIdx.x >> 5;
    const int row  = blockIdx.x * 8 + wid;
    if (row >= N) return;

    const float* x  = X + (size_t)row * NODE_DIM;
    const int d0 = lane;
    const int d1 = lane + 32;
    const float* w0 = W + (size_t)d0 * NODE_DIM;
    const float* w1 = W + (size_t)d1 * NODE_DIM;

    float a0 = bias[d0];
    float a1 = bias[d1];
#pragma unroll 8
    for (int k = 0; k < NODE_DIM; k += 4) {
        float4 xv  = *(const float4*)(x  + k);
        float4 wv0 = *(const float4*)(w0 + k);
        float4 wv1 = *(const float4*)(w1 + k);
        a0 = fmaf(xv.x, wv0.x, a0); a0 = fmaf(xv.y, wv0.y, a0);
        a0 = fmaf(xv.z, wv0.z, a0); a0 = fmaf(xv.w, wv0.w, a0);
        a1 = fmaf(xv.x, wv1.x, a1); a1 = fmaf(xv.y, wv1.y, a1);
        a1 = fmaf(xv.z, wv1.z, a1); a1 = fmaf(xv.w, wv1.w, a1);
    }
    a0 = fmaxf(a0, 0.0f);
    a1 = fmaxf(a1, 0.0f);

    // bf16 hi/lo split (x ~= hi + lo), both RNE
    unsigned short h0 = f32_to_bf16_rne(a0);
    unsigned short h1 = f32_to_bf16_rne(a1);
    unsigned short l0 = f32_to_bf16_rne(a0 - bf16_bits_to_f32(h0));
    unsigned short l1 = f32_to_bf16_rne(a1 - bf16_bits_to_f32(h1));

    const size_t base = (size_t)row * HIDDEN_DIM;
    hhi[base + d0] = h0;
    hhi[base + d1] = h1;
    hlo[base + d0] = l0;
    hlo[base + d1] = l1;

    // wave32 reduction of ||h||^2
    float loc = fmaf(a0, a0, a1 * a1);
#pragma unroll
    for (int m = 16; m >= 1; m >>= 1)
        loc += __shfl_xor(loc, m, 32);
    if (lane == 0) sq[row] = loc;
}

// ---------------------------------------------------------------------------
// Kernel 2: one 16x16 Gram tile per wave via v_wmma_f32_16x16x32_bf16.
// Fragment loads use the same contiguous K order for A(16x32) and B(32x16):
//   A: lane L    -> row i0+(L&15), 16 contiguous K starting at 16*(L>=16)
//   B: lane L    -> row j0+(L&15), 16 contiguous K starting at 16*(L>=16)
// C/D layout: VGPR v, lanes 0-15 -> (M=v, N=lane); lanes 16-31 -> (M=v+8).
// Interior tiles (j0 >= i0+16) take a branch-free epilogue; only diagonal
// tiles use the masked path. All outputs written with non-temporal stores.
// ---------------------------------------------------------------------------
__global__ void pair_kernel(const unsigned short* __restrict__ hhi,
                            const unsigned short* __restrict__ hlo,
                            const float* __restrict__ sq,
                            const float* __restrict__ thrp,
                            float* __restrict__ probs,
                            long long* __restrict__ out_ii,
                            long long* __restrict__ out_jj,
                            int N) {
    const int lane = threadIdx.x & 31;
    const int wid  = threadIdx.x >> 5;
    const int i0 = (blockIdx.y * 8 + wid) * 16;
    const int j0 = blockIdx.x * 16;
    // wave-uniform early exit: tile entirely below diagonal (no j > i)
    if (i0 >= N || (j0 + 15) <= i0) return;

    const int r     = lane & 15;
    const int upper = (lane >> 4) & 1;       // 0 for lanes 0-15, 1 for 16-31
    const int ka    = upper * 16;            // A K-offset within 32-block
    const int kbq   = upper * 16;            // B K-offset within 32-block

    const unsigned short* arow_hi = hhi + (size_t)(i0 + r) * HIDDEN_DIM;
    const unsigned short* arow_lo = hlo + (size_t)(i0 + r) * HIDDEN_DIM;
    const unsigned short* brow_hi = hhi + (size_t)(j0 + r) * HIDDEN_DIM;
    const unsigned short* brow_lo = hlo + (size_t)(j0 + r) * HIDDEN_DIM;

    v8f c = {};
#pragma unroll
    for (int kb = 0; kb < 2; ++kb) {         // two K=32 blocks cover K=64
        const int kofs = kb * 32;
        V16U ah, al;
        ah.h[0] = *(const v8bf*)(const void*)(arow_hi + kofs + ka);
        ah.h[1] = *(const v8bf*)(const void*)(arow_hi + kofs + ka + 8);
        al.h[0] = *(const v8bf*)(const void*)(arow_lo + kofs + ka);
        al.h[1] = *(const v8bf*)(const void*)(arow_lo + kofs + ka + 8);
        v16bf bh = *(const v16bf*)(const void*)(brow_hi + kofs + kbq);
        v16bf bl = *(const v16bf*)(const void*)(brow_lo + kofs + kbq);

        // split-compensated product: hi*hi + hi*lo + lo*hi  (~fp32 accuracy)
        c = __builtin_amdgcn_wmma_f32_16x16x32_bf16(false, ah.v, false, bh,
                                                    (short)0, c, false, false);
        c = __builtin_amdgcn_wmma_f32_16x16x32_bf16(false, ah.v, false, bl,
                                                    (short)0, c, false, false);
        c = __builtin_amdgcn_wmma_f32_16x16x32_bf16(false, al.v, false, bh,
                                                    (short)0, c, false, false);
    }

    const float thr   = *thrp;
    const float sqj   = sq[j0 + r];
    const int   mbase = upper * 8;
    const int   j     = j0 + r;

    // preload all 8 row norms up front (clause-able, outside divergence)
    float sqi[8];
#pragma unroll
    for (int v = 0; v < 8; ++v)
        sqi[v] = sq[i0 + mbase + v];

    // N = 8192 -> pair count and all index math fit in u32
    const unsigned Nu = (unsigned)N;

    if (j0 >= i0 + 16) {
        // interior tile: every (i,j) has j > i -> branch-free epilogue
#pragma unroll
        for (int v = 0; v < 8; ++v) {
            const int i = i0 + mbase + v;
            const float p = edge_prob(c[v], sqi[v], sqj, thr);
            const unsigned iu  = (unsigned)i;
            const unsigned idx = iu * Nu - (iu * (iu + 1u)) / 2u
                               + (unsigned)(j - i - 1);
            __builtin_nontemporal_store(p, probs + idx);
            __builtin_nontemporal_store((long long)i, out_ii + idx);
            __builtin_nontemporal_store((long long)j, out_jj + idx);
        }
    } else {
        // diagonal tile: mask j > i
#pragma unroll
        for (int v = 0; v < 8; ++v) {
            const int i = i0 + mbase + v;
            if (j > i) {
                const float p = edge_prob(c[v], sqi[v], sqj, thr);
                const unsigned iu  = (unsigned)i;
                const unsigned idx = iu * Nu - (iu * (iu + 1u)) / 2u
                                   + (unsigned)(j - i - 1);
                __builtin_nontemporal_store(p, probs + idx);
                __builtin_nontemporal_store((long long)i, out_ii + idx);
                __builtin_nontemporal_store((long long)j, out_jj + idx);
            }
        }
    }
}

// ---------------------------------------------------------------------------
extern "C" void kernel_launch(void* const* d_in, const int* in_sizes, int n_in,
                              void* d_out, int out_size, void* d_ws, size_t ws_size,
                              hipStream_t stream) {
    const float* X   = (const float*)d_in[0];   // [N, 128]
    const float* W   = (const float*)d_in[1];   // [64, 128]
    const float* b   = (const float*)d_in[2];   // [64]
    const float* thr = (const float*)d_in[3];   // scalar

    const int N = in_sizes[0] / NODE_DIM;

    // workspace layout: hhi [N*64] bf16, hlo [N*64] bf16, sq [N] f32
    unsigned short* hhi = (unsigned short*)d_ws;
    unsigned short* hlo = hhi + (size_t)N * HIDDEN_DIM;
    float*          sqv = (float*)(hlo + (size_t)N * HIDDEN_DIM);

    // output layout: probs [P] f32, then pair_index [2, P] int64
    const long long P = (long long)N * (N - 1) / 2;
    float*     probs  = (float*)d_out;
    long long* out_ii = (long long*)(probs + P);
    long long* out_jj = out_ii + P;

    // K1: 8 waves/block, one row per wave
    dim3 g1((N + 7) / 8);
    encode_kernel<<<g1, 256, 0, stream>>>(X, W, b, hhi, hlo, sqv, N);

    // K2: 8 waves/block, one 16x16 tile per wave; y covers i-tiles, x j-tiles
    const int tiles = (N + 15) / 16;
    dim3 g2(tiles, (tiles + 7) / 8);
    pair_kernel<<<g2, 256, 0, stream>>>(hhi, hlo, sqv, thr,
                                        probs, out_ii, out_jj, N);
}
